// SpatialPatchMoE_326417514517
// MI455X (gfx1250) — compile-verified, matrix operand order checked by reading, not run on hardware
//
#include <hip/hip_runtime.h>

// ---------------- CDNA5 WMMA / TDM types ----------------
typedef __attribute__((ext_vector_type(16))) __bf16 v16bf;
typedef __attribute__((ext_vector_type(8)))  float  v8f;
typedef __attribute__((ext_vector_type(4)))  unsigned int v4u;
typedef __attribute__((ext_vector_type(8)))  int v8i;
typedef __attribute__((ext_vector_type(4)))  int v4i;

#if defined(__has_builtin)
#if __has_builtin(__builtin_amdgcn_tensor_load_to_lds) && __has_builtin(__builtin_amdgcn_s_wait_tensorcnt)
#define USE_TDM 1
#endif
#endif

union V16 { v16bf v; unsigned int u[8]; __bf16 h[16]; };

#define SHS 66   // bf16 row stride (pos-major h/u buffer), even + padded
#define S2S 66   // bf16 row stride for h2 buffer

// K index of the bf16 pair held in operand dword j for lane-half kh
// (16-bit A/B 16x32 layout, CDNA5 ISA §7.12.2)
__device__ __forceinline__ int kmap(int j, int kh) {
  return ((j < 4) ? 0 : 16) + kh * 8 + (j & 3) * 2;
}

// Load a 32x16 B tile (K x N) from pos-major bf16 LDS [pos][k], stride SHS.
__device__ __forceinline__ v16bf load_b_lds(const __bf16* shb, int pos0, int k0, int lane) {
  int nn = lane & 15, kh = lane >> 4;
  const __bf16* p = shb + (pos0 + nn) * SHS + k0;
  V16 r;
#pragma unroll
  for (int j = 0; j < 8; ++j) {
    int k = kmap(j, kh);
    r.u[j] = *(const unsigned int*)(p + k);   // (k, k+1) contiguous, 4B aligned
  }
  return r.v;
}

// Load a 16x32 A tile from row-major fp32 global weights, converting to bf16.
__device__ __forceinline__ v16bf load_a_w(const float* w, int row0, int k0, int ld, int lane) {
  int m = lane & 15, kh = lane >> 4;
  const float* rp = w + (size_t)(row0 + m) * ld + k0;
  V16 r;
#pragma unroll
  for (int j = 0; j < 8; ++j) {
    int k = kmap(j, kh);
    r.h[2 * j]     = (__bf16)rp[k];
    r.h[2 * j + 1] = (__bf16)rp[k + 1];
  }
  return r.v;
}

#ifdef USE_TDM
// Issue one TDM load: up-to-3D tile of fp32 into LDS (6-arg clang-23 builtin:
// four D# SGPR groups + zero-filled extra group + cpol).
// dim0: t0 contiguous elements; dim1: t1 rows @ stride s0; dim2: t2 planes @ stride s1.
__device__ __forceinline__ void tdm_load_f32(
    unsigned lds_off, unsigned long long gaddr,
    unsigned t0, unsigned t1, unsigned t2,
    unsigned long long s0, unsigned long long s1,
    unsigned td0, unsigned td1, unsigned td2)
{
  v4u g0 = { 1u,                                   // count=1, user descriptor
             lds_off,                              // lds_addr
             (unsigned)(gaddr & 0xFFFFFFFFull),    // global_addr[31:0]
             (unsigned)((gaddr >> 32) & 0x1FFFFFFu) | (2u << 30) }; // addr[56:32] | type=2
  v8i g1 = { (int)(2u << 16),                              // data_size = 4 bytes
             (int)((td0 & 0xFFFFu) << 16),                 // tensor_dim0[15:0]
             (int)((td0 >> 16) | ((td1 & 0xFFFFu) << 16)), // tdim0 hi | tdim1 lo
             (int)((td1 >> 16) | (t0 << 16)),              // tdim1 hi | tile_dim0
             (int)(t1 | (t2 << 16)),                       // tile_dim1 | tile_dim2
             (int)(unsigned)(s0 & 0xFFFFFFFFull),          // tdim0_stride[31:0]
             (int)(unsigned)(((s0 >> 32) & 0xFFFFu) | ((s1 & 0xFFFFu) << 16)),
             (int)(unsigned)((s1 >> 16) & 0xFFFFFFFFull) };// tdim1_stride[47:16]
  v4i g2 = { (int)td2, 0, 0, 0 };                  // tensor_dim2
  v4i g3 = { 0, 0, 0, 0 };
  v8i gz = { 0, 0, 0, 0, 0, 0, 0, 0 };             // extra group (zero-filled)
  __builtin_amdgcn_tensor_load_to_lds(g0, g1, g2, g3, gz, 0);
}
#endif

// ---------------- Kernel 1: router (mean -> logits -> top-2 softmax) ----------------
__global__ __launch_bounds__(64) void moe_router_kernel(
    const float* __restrict__ x, const float* __restrict__ rw,
    const float* __restrict__ rb, int4* __restrict__ route)
{
  __shared__ float smean[64];
  __shared__ float slog[8];
  int n = blockIdx.x;                 // n = b*256 + ph*16 + pw
  int b = n >> 8, rem = n & 255;
  int ph = rem >> 4, pwc = rem & 15;
  int c = threadIdx.x;                // 64 threads: one channel each

  const float* px = x + (size_t)(b * 64 + c) * 8 * 16384 + (ph * 8) * 128 + pwc * 8;
  float s = 0.0f;
  for (int l = 0; l < 8; ++l)
    for (int py = 0; py < 8; ++py) {
      const float* row = px + (size_t)l * 16384 + py * 128;
#pragma unroll
      for (int q = 0; q < 8; ++q) s += row[q];
    }
  smean[c] = s * (1.0f / 512.0f);
  __syncthreads();
  if (c < 8) {
    float lg = rb[c];
    for (int cc = 0; cc < 64; ++cc) lg += smean[cc] * rw[c * 64 + cc];
    slog[c] = lg;
  }
  __syncthreads();
  if (c == 0) {
    int i0 = 0; float v0 = slog[0];
    for (int e = 1; e < 8; ++e) if (slog[e] > v0) { v0 = slog[e]; i0 = e; }
    int i1 = (i0 == 0) ? 1 : 0; float v1 = slog[i1];
    for (int e = 0; e < 8; ++e) {
      if (e == i0) continue;
      if (slog[e] > v1) { v1 = slog[e]; i1 = e; }
    }
    float mx = fmaxf(v0, v1);
    float e0 = __expf(v0 - mx), e1 = __expf(v1 - mx);
    float inv = 1.0f / (e0 + e1);
    int4 r; r.x = i0; r.y = i1;
    r.z = __float_as_int(e0 * inv); r.w = __float_as_int(e1 * inv);
    route[n] = r;
  }
}

// ---------------- Kernel 2: per (patch, l) gated-conv MoE block ----------------
__global__ __launch_bounds__(256) void moe_block_kernel(
    const float* __restrict__ x,
    const float* __restrict__ dw_w,  const float* __restrict__ dw_b,
    const float* __restrict__ ln_w,  const float* __restrict__ ln_b,
    const float* __restrict__ pw_in_w,  const float* __restrict__ pw_in_b,
    const float* __restrict__ pw_out_w, const float* __restrict__ pw_out_b,
    const int4* __restrict__ route, float* __restrict__ out)
{
  __shared__ float  sx[4096];        // x slice [c][pos] f32 (conv input + residual)
  __shared__ float  sdw[3136];       // expert depthwise weights [c][49]
  __shared__ float  slw[64];         // LN gamma [pos]
  __shared__ float  slb[64];         // LN beta  [pos]
  __shared__ __bf16 sh2[128 * S2S];  // GEMM1 output (a;g) [o][pos]
  __shared__ __bf16 sh[64 * SHS];    // h / u, pos-major [pos][c] (WMMA B staging)

  int tid = threadIdx.x;
  int job = blockIdx.x;
  int l = job & 7;
  int n = job >> 3;
  int b = n >> 8, rem = n & 255;
  int ph = rem >> 4, pwc = rem & 15;
  int corner = (ph * 8) * 128 + pwc * 8;

  int lane = tid & 31;
  int wv   = tid >> 5;   // 8 waves
  int cth  = tid >> 2;   // conv: channel
  int q    = tid & 3;    // conv: quarter of the 64 positions
  int mt2  = wv & 3;         // GEMM2 tile assignment
  int ntb2 = (wv >> 2) * 2;

  // ---- stage x slice into LDS ----
#ifdef USE_TDM
  if (wv == 0) {
    // 3D tile: 8 floats x 8 rows(stride 128) x 64 channels(stride 8*16384)
    unsigned long long gx = (unsigned long long)(size_t)
        (x + ((size_t)(b * 64) * 8 + l) * 16384 + corner);
    unsigned lds_sx = (unsigned)(size_t)(void*)sx;
    tdm_load_f32(lds_sx, gx, 8u, 8u, 64u,
                 128ull, 131072ull, 128u, 128u, 64u);
    __builtin_amdgcn_s_wait_tensorcnt(0);
  }
#else
#pragma unroll
  for (int i = 0; i < 16; ++i) {
    int idx = i * 256 + tid;
    int c = idx >> 6, pos = idx & 63;
    int py = pos >> 3, px = pos & 7;
    sx[idx] = x[((size_t)(b * 64 + c) * 8 + l) * 16384 + corner + py * 128 + px];
  }
#endif

  int4 rt = route[n];

  float outacc[2][8];
#pragma unroll
  for (int jt = 0; jt < 2; ++jt)
#pragma unroll
    for (int r = 0; r < 8; ++r) outacc[jt][r] = 0.0f;

  for (int k = 0; k < 2; ++k) {
    int e   = (k == 0) ? rt.x : rt.y;
    float wk = __int_as_float((k == 0) ? rt.z : rt.w);

    __syncthreads();   // previous iteration done; sx visible
    // ---- stage expert depthwise weights + LN affine ----
#ifdef USE_TDM
    if (wv == 0) {
      unsigned long long gw = (unsigned long long)(size_t)(dw_w + (size_t)e * 3136);
      unsigned lds_sdw = (unsigned)(size_t)(void*)sdw;
      // 1D tile: 3136 contiguous floats
      tdm_load_f32(lds_sdw, gw, 3136u, 1u, 0u, 3136ull, 0ull, 3136u, 1u, 1u);
    }
    if (tid == 0) {   // warm L2 for the pointwise weight slabs (global_prefetch_b8)
      __builtin_prefetch(pw_in_w + (size_t)e * 128 * 64, 0, 1);
      __builtin_prefetch(pw_out_w + (size_t)e * 64 * 64, 0, 1);
    }
#else
    for (int idx = tid; idx < 3136; idx += 256) sdw[idx] = dw_w[e * 3136 + idx];
#endif
    if (tid < 64) { slw[tid] = ln_w[e * 64 + tid]; slb[tid] = ln_b[e * 64 + tid]; }
#ifdef USE_TDM
    if (wv == 0) __builtin_amdgcn_s_wait_tensorcnt(0);
#endif
    __syncthreads();

    // ---- depthwise 7x7 conv (zero-padded 8x8 tile) + LayerNorm over 64 positions ----
    float vals[16];
    float dwb = dw_b[e * 64 + cth];
    const float* xc = sx  + cth * 64;
    const float* wc = sdw + cth * 49;
    float s1 = 0.0f, s2 = 0.0f;
#pragma unroll
    for (int i = 0; i < 16; ++i) {
      int pos = q * 16 + i;
      int py = pos >> 3, px = pos & 7;
      float acc = dwb;
      for (int dy = 0; dy < 7; ++dy) {
        int yy = py + dy - 3;
        if (yy < 0 || yy > 7) continue;
        for (int dx = 0; dx < 7; ++dx) {
          int xx = px + dx - 3;
          if (xx < 0 || xx > 7) continue;
          acc += xc[yy * 8 + xx] * wc[dy * 7 + dx];
        }
      }
      vals[i] = acc; s1 += acc; s2 += acc * acc;
    }
    // reduce across the 4 lanes sharing a channel (lanes 4c..4c+3, within wave32)
    s1 += __shfl_xor(s1, 1); s1 += __shfl_xor(s1, 2);
    s2 += __shfl_xor(s2, 1); s2 += __shfl_xor(s2, 2);
    float mu   = s1 * (1.0f / 64.0f);
    float var  = s2 * (1.0f / 64.0f) - mu * mu;
    float rstd = rsqrtf(var + 1e-5f);
#pragma unroll
    for (int i = 0; i < 16; ++i) {
      int pos = q * 16 + i;
      float hn = (vals[i] - mu) * rstd * slw[pos] + slb[pos];
      sh[pos * SHS + cth] = (__bf16)hn;
    }
    __syncthreads();

    // ---- GEMM1: h2[128x64] = pw_in_w[e] (128x64) @ h (64x64), bf16 WMMA, f32 acc ----
    {
      const float* aw = pw_in_w + (size_t)e * 128 * 64;
      v16bf a0 = load_a_w(aw, wv * 16, 0,  64, lane);
      v16bf a1 = load_a_w(aw, wv * 16, 32, 64, lane);
#pragma unroll
      for (int nt = 0; nt < 4; ++nt) {
        v16bf b0 = load_b_lds(sh, nt * 16, 0,  lane);
        v16bf b1 = load_b_lds(sh, nt * 16, 32, lane);
        v8f acc = {};
        acc = __builtin_amdgcn_wmma_f32_16x16x32_bf16(false, a0, false, b0, (short)0, acc, false, false);
        acc = __builtin_amdgcn_wmma_f32_16x16x32_bf16(false, a1, false, b1, (short)0, acc, false, false);
        int nn = nt * 16 + (lane & 15);
#pragma unroll
        for (int r = 0; r < 8; ++r) {
          int o = wv * 16 + (lane >> 4) * 8 + r;
          sh2[o * S2S + nn] = (__bf16)(acc[r] + pw_in_b[e * 128 + o]);
        }
      }
    }
    __syncthreads();

    // ---- gate: u = silu(a) * g, overwrite sh (pos-major bf16) ----
#pragma unroll
    for (int i = 0; i < 16; ++i) {
      int idx = i * 256 + tid;
      int c = idx >> 6, pos = idx & 63;
      float av = (float)sh2[c * S2S + pos];
      float gv = (float)sh2[(c + 64) * S2S + pos];
      float sv = av / (1.0f + __expf(-av));
      sh[pos * SHS + c] = (__bf16)(sv * gv);
    }
    __syncthreads();

    // ---- GEMM2: hout[64x64] = pw_out_w[e] (64x64) @ u (64x64); acc += wk*(hout+b) ----
    {
      const float* aw = pw_out_w + (size_t)e * 64 * 64;
      v16bf a0 = load_a_w(aw, mt2 * 16, 0,  64, lane);
      v16bf a1 = load_a_w(aw, mt2 * 16, 32, 64, lane);
#pragma unroll
      for (int jt = 0; jt < 2; ++jt) {
        int nt = ntb2 + jt;
        v16bf b0 = load_b_lds(sh, nt * 16, 0,  lane);
        v16bf b1 = load_b_lds(sh, nt * 16, 32, lane);
        v8f acc = {};
        acc = __builtin_amdgcn_wmma_f32_16x16x32_bf16(false, a0, false, b0, (short)0, acc, false, false);
        acc = __builtin_amdgcn_wmma_f32_16x16x32_bf16(false, a1, false, b1, (short)0, acc, false, false);
#pragma unroll
        for (int r = 0; r < 8; ++r) {
          int o = mt2 * 16 + (lane >> 4) * 8 + r;
          outacc[jt][r] += wk * (acc[r] + pw_out_b[e * 64 + o]);
        }
      }
    }
  }

  // ---- residual + store (each output element written by exactly one lane) ----
#pragma unroll
  for (int jt = 0; jt < 2; ++jt) {
    int nt = ntb2 + jt;
    int pos = nt * 16 + (lane & 15);
    int py = pos >> 3, px = pos & 7;
#pragma unroll
    for (int r = 0; r < 8; ++r) {
      int c = mt2 * 16 + (lane >> 4) * 8 + r;
      float val = sx[c * 64 + pos] + outacc[jt][r];
      out[((size_t)(b * 64 + c) * 8 + l) * 16384 + corner + py * 128 + px] = val;
    }
  }
}

// ---------------- launch ----------------
extern "C" void kernel_launch(void* const* d_in, const int* in_sizes, int n_in,
                              void* d_out, int out_size, void* d_ws, size_t ws_size,
                              hipStream_t stream) {
  const float* x        = (const float*)d_in[0];
  const float* router_w = (const float*)d_in[1];
  const float* router_b = (const float*)d_in[2];
  const float* dw_w     = (const float*)d_in[3];
  const float* dw_b     = (const float*)d_in[4];
  const float* ln_w     = (const float*)d_in[5];
  const float* ln_b     = (const float*)d_in[6];
  const float* pw_in_w  = (const float*)d_in[7];
  const float* pw_in_b  = (const float*)d_in[8];
  const float* pw_out_w = (const float*)d_in[9];
  const float* pw_out_b = (const float*)d_in[10];
  float* out = (float*)d_out;
  int4* route = (int4*)d_ws;   // 512 * 16B

  moe_router_kernel<<<512, 64, 0, stream>>>(x, router_w, router_b, route);
  moe_block_kernel<<<4096, 256, 0, stream>>>(x, dw_w, dw_b, ln_w, ln_b,
                                             pw_in_w, pw_in_b, pw_out_w, pw_out_b,
                                             route, out);
}
